// CoXtLayer_19533511262755
// MI455X (gfx1250) — matrix operand order repaired and checked by reading, not compiled, not run on hardware
//
#include <hip/hip_runtime.h>

// ---------------------------------------------------------------------------
// CDNA5 (gfx1250) implementation of the CotNet-style layer.
// GEMM stages: V_WMMA_F32_16X16X32_BF16 (bf16 in, fp32 accum).
// B-tile staging: Tensor Data Mover (tensor_load_to_lds, 6-arg clang-23 form),
// double-buffered, tracked with TENSORcnt.
// ---------------------------------------------------------------------------

typedef __attribute__((ext_vector_type(16))) __bf16 v16bf;
typedef __attribute__((ext_vector_type(8)))  float  v8f;
typedef __attribute__((ext_vector_type(4)))  unsigned v4u;
typedef __attribute__((ext_vector_type(8)))  int      v8i;
typedef __attribute__((ext_vector_type(4)))  int      v4i;

union BF { v16bf v; unsigned d[8]; };

static constexpr int H_ = 56, W_ = 56, HW_ = 3136, B_ = 16, C_ = 256;

#if __has_builtin(__builtin_amdgcn_tensor_load_to_lds) && \
    __has_builtin(__builtin_amdgcn_s_wait_tensorcnt)
#define USE_TDM 1
#endif

__device__ __forceinline__ unsigned short f2bf(float f) {
    unsigned u = __float_as_uint(f);
    unsigned r = u + 0x7FFFu + ((u >> 16) & 1u);
    return (unsigned short)(r >> 16);
}

// A fragment: 16(M) x 32(K) bf16 from LDS; row stride in u32 units, pointer
// pre-offset to the K-chunk (16 u32 = 32 bf16 per chunk).
__device__ __forceinline__ void loadA_s(BF& a, const unsigned* ldsA, int mBase,
                                        int lane, int strideU32) {
    int lo = lane & 15, hi = lane >> 4;
    const unsigned* row = ldsA + (size_t)(mBase + lo) * strideU32;
#pragma unroll
    for (int j = 0; j < 8; ++j) {
        a.d[j] = row[hi * 8 + j];           // K 0..15 (lo half) / 16..31 (hi half)
    }
}

// B fragment: 32(K) x 16(N) bf16, LDS tile layout [col][32] (col stride 16 u32)
__device__ __forceinline__ void loadB(BF& b, const unsigned* ldsB, int nBase, int lane) {
    int lo = lane & 15, hi = lane >> 4;
    const unsigned* col = ldsB + (size_t)(nBase + lo) * 16;
#pragma unroll
    for (int j = 0; j < 8; ++j)
        b.d[j] = col[hi * 8 + j];
}

__device__ __forceinline__ v8f wmma_bf16(const BF& a, const BF& b, v8f c) {
    return __builtin_amdgcn_wmma_f32_16x16x32_bf16(false, a.v, false, b.v,
                                                   (short)0, c, false, false);
}

#ifdef USE_TDM
// Issue one TDM 2D tile load: rows x cols fp32, row stride rowStrideElems,
// into LDS at byte offset ldsByteOff. Descriptor per CDNA5 ISA D# layout.
__device__ __forceinline__ void tdm_load_2d(unsigned ldsByteOff, const float* gaddr,
                                            int rows, int cols, int rowStrideElems) {
    unsigned long long ga = (unsigned long long)(size_t)gaddr;
    v4u g0;
    g0[0] = 1u;                                             // count=1 (valid D#)
    g0[1] = ldsByteOff;                                     // lds_addr
    g0[2] = (unsigned)(ga & 0xFFFFFFFFu);                   // global_addr[31:0]
    g0[3] = (unsigned)((ga >> 32) & 0x01FFFFFFu) | (2u << 30);  // [56:32] | type=2
    const unsigned td0 = 1u << 30, td1 = 1u << 20;          // large => no OOB clamp
    unsigned long long s0 = (unsigned long long)rowStrideElems;
    v8i g1;
    g1[0] = (int)(2u << 16);                                // data_size = 4B
    g1[1] = (int)((td0 & 0xFFFFu) << 16);                   // tensor_dim0[15:0]
    g1[2] = (int)(((td0 >> 16) & 0xFFFFu) | ((td1 & 0xFFFFu) << 16));
    g1[3] = (int)(((td1 >> 16) & 0xFFFFu) | ((unsigned)cols << 16)); // tile_dim0
    g1[4] = (int)((unsigned)rows & 0xFFFFu);                // tile_dim1, tile_dim2=0
    g1[5] = (int)(unsigned)(s0 & 0xFFFFFFFFu);              // dim0_stride[31:0]
    g1[6] = (int)(unsigned)((s0 >> 32) & 0xFFFFu);          // dim0_stride[47:32]
    g1[7] = 0;
    v4i z4 = {0, 0, 0, 0};
    v8i z8 = {0, 0, 0, 0, 0, 0, 0, 0};
    __builtin_amdgcn_tensor_load_to_lds(g0, g1, z4, z4, z8, 0);
}
#endif

// ---------------------------------------------------------------------------
// Kernel 1: grouped 3x3 conv (groups=8, 32->32 per group) + BN + ReLU -> k
// ---------------------------------------------------------------------------
__global__ void ke_conv3x3(const float* __restrict__ x, const float* __restrict__ kw,
                           const float* __restrict__ gg, const float* __restrict__ gb,
                           float* __restrict__ kout) {
    const int tileH = blockIdx.x / 7, tileW = blockIdx.x % 7;
    const int g = blockIdx.y, b = blockIdx.z;
    const int tid = threadIdx.x;

    __shared__ alignas(16) unsigned short sIn[100 * 32];      // 10x10 halo x 32ch
    __shared__ alignas(16) unsigned short sW[9 * 32 * 32];    // [tap][co][ci]

    for (int idx = tid; idx < 9216; idx += 256) {
        int t = idx >> 10, rem = idx & 1023;
        int m = rem >> 5, ci = rem & 31;
        sW[idx] = f2bf(kw[(((size_t)(g * 32 + m) * 32) + ci) * 9 + t]);
    }
    for (int idx = tid; idx < 3200; idx += 256) {
        int ch = idx / 100, sp = idx % 100;
        int r = sp / 10, cc = sp % 10;
        int gh = tileH * 8 + r - 1, gw = tileW * 8 + cc - 1;
        float val = 0.f;
        if (gh >= 0 && gh < H_ && gw >= 0 && gw < W_)
            val = x[((size_t)b * C_ + (g * 32 + ch)) * HW_ + gh * W_ + gw];
        sIn[sp * 32 + ch] = f2bf(val);
    }
    __syncthreads();

    const int wid = tid >> 5, lane = tid & 31, lo = lane & 15, hi = lane >> 4;
    const int mBase = (wid >> 2) * 16, nBase = (wid & 3) * 16;
    const unsigned* uW  = (const unsigned*)sW;
    const unsigned* uIn = (const unsigned*)sIn;

    v8f acc = {0.f, 0.f, 0.f, 0.f, 0.f, 0.f, 0.f, 0.f};
#pragma unroll
    for (int t = 0; t < 9; ++t) {
        int dy = t / 3, dx = t % 3;
        BF a, bf;
        loadA_s(a, uW + t * 512, mBase, lane, 16);
        int n = nBase + lo;
        int sp = (n / 8 + dy) * 10 + (n % 8 + dx);
        const unsigned* col = uIn + sp * 16;
#pragma unroll
        for (int j = 0; j < 8; ++j) bf.d[j] = col[hi * 8 + j];
        acc = wmma_bf16(a, bf, acc);
    }

    const float BNS = rsqrtf(1.f + 1e-5f);
#pragma unroll
    for (int r = 0; r < 8; ++r) {
        int m = mBase + hi * 8 + r;
        int o = g * 32 + m;
        int n = nBase + lo;
        int gh = tileH * 8 + n / 8, gw = tileW * 8 + (n & 7);
        float val = fmaxf(0.f, acc[r] * (gg[o] * BNS) + gb[o]);
        kout[((size_t)b * C_ + o) * HW_ + gh * W_ + gw] = val;
    }
}

// ---------------------------------------------------------------------------
// Kernels 2/3/4: grouped 1x1 convs as WMMA GEMMs with TDM-staged B tiles.
// MODE 0: em1 (qk interleave of x,k; 256->64 per group; BN+ReLU)
// MODE 1: em2 (e1; 64->144 per group; +bias)
// MODE 2: c1  (x; 128->128 per group; BN)
// Block 256 thr (8 waves, 4(M) x 2(N)); block tile 64co x 64px; K chunk 32.
// ---------------------------------------------------------------------------
template <int CIN_G, int COUT_G, int MODE>
__global__ void conv1x1_wmma(const float* __restrict__ inA, const float* __restrict__ inB,
                             const float* __restrict__ wgt,
                             const float* __restrict__ p0, const float* __restrict__ p1,
                             float* __restrict__ outp) {
    const int pxBase = blockIdx.x * 64;
    const int mT = blockIdx.y;
    const int b = blockIdx.z >> 1, g = blockIdx.z & 1;
    const int tid = threadIdx.x;
    const int wid = tid >> 5, lane = tid & 31, lo = lane & 15, hi = lane >> 4;
    const int mBase = (wid >> 1) * 16, waveN = wid & 1;
    const int COUT_TOT = 2 * COUT_G;
    constexpr int NK = CIN_G / 32;

    __shared__ alignas(16) unsigned short sA[64 * CIN_G];   // full A tile (bf16)
    __shared__ alignas(16) unsigned short sB[64 * 32];      // B chunk, [px][ch] bf16
#ifdef USE_TDM
    __shared__ alignas(16) float sF[2][32 * 64];            // TDM fp32 double buffer
#endif

    // Stage the full weight tile once (rows past COUT_G zero-padded).
    for (int idx = tid; idx < 64 * CIN_G; idx += 256) {
        int row = idx / CIN_G, kk = idx % CIN_G;
        int orow = mT * 64 + row;
        float wv = 0.f;
        if (orow < COUT_G)
            wv = wgt[((size_t)(g * COUT_G + orow)) * CIN_G + kk];
        sA[idx] = f2bf(wv);
    }

    v8f acc0 = {0.f, 0.f, 0.f, 0.f, 0.f, 0.f, 0.f, 0.f};
    v8f acc1 = {0.f, 0.f, 0.f, 0.f, 0.f, 0.f, 0.f, 0.f};

#ifdef USE_TDM
    constexpr int NDESC = (MODE == 0) ? 2 : 1;
    const unsigned sFbase = (unsigned)(unsigned long long)(size_t)&sF[0][0];
    auto issueChunk = [&](int kc, int buf) {
        unsigned off = sFbase + (unsigned)buf * (32 * 64 * 4);
        if (MODE == 0) {   // 16 rows from x, 16 rows from k (interleave resolved later)
            int c0 = g * 128 + kc * 16;
            tdm_load_2d(off,               inA + ((size_t)b * 256 + c0) * HW_ + pxBase,
                        16, 64, HW_);
            tdm_load_2d(off + 16 * 64 * 4, inB + ((size_t)b * 256 + c0) * HW_ + pxBase,
                        16, 64, HW_);
        } else if (MODE == 1) {
            tdm_load_2d(off, inA + ((size_t)b * 128 + g * 64 + kc * 32) * HW_ + pxBase,
                        32, 64, HW_);
        } else {
            tdm_load_2d(off, inA + ((size_t)b * 256 + g * 128 + kc * 32) * HW_ + pxBase,
                        32, 64, HW_);
        }
    };
    if (wid == 0) issueChunk(0, 0);
#endif

    for (int kc = 0; kc < NK; ++kc) {
#ifdef USE_TDM
        const int cur = kc & 1;
        if (wid == 0) {
            if (kc + 1 < NK) {
                issueChunk(kc + 1, cur ^ 1);                 // prefetch next chunk
                __builtin_amdgcn_s_wait_tensorcnt(NDESC);    // current chunk done
            } else {
                __builtin_amdgcn_s_wait_tensorcnt(0);
            }
        }
        __syncthreads();   // TDM data visible; prev frag reads done before sB overwrite
        // Convert fp32 tile -> bf16 [px][ch]; fold qk interleave for MODE 0.
        {
            const float* fb = &sF[cur][0];
            for (int idx = tid; idx < 2048; idx += 256) {
                int px = idx & 63, ch = idx >> 6;
                int row = (MODE == 0) ? (((ch & 1) ? 16 : 0) + (ch >> 1)) : ch;
                sB[px * 32 + ch] = f2bf(fb[row * 64 + px]);
            }
        }
        __syncthreads();
#else
        __syncthreads();
        for (int idx = tid; idx < 2048; idx += 256) {
            int ch = idx >> 6, px = idx & 63;
            int j = kc * 32 + ch;
            float val;
            if (MODE == 0) {
                int c = g * 128 + (j >> 1);
                const float* src = (j & 1) ? inB : inA;
                val = src[((size_t)b * 256 + c) * HW_ + pxBase + px];
            } else if (MODE == 1) {
                val = inA[((size_t)b * 128 + g * 64 + j) * HW_ + pxBase + px];
            } else {
                val = inA[((size_t)b * 256 + g * 128 + j) * HW_ + pxBase + px];
            }
            sB[px * 32 + ch] = f2bf(val);
        }
        __syncthreads();
#endif
        BF a, b0, b1;
        loadA_s(a, (const unsigned*)sA + kc * 16, mBase, lane, CIN_G / 2);
        loadB(b0, (const unsigned*)sB, waveN * 32, lane);
        loadB(b1, (const unsigned*)sB, waveN * 32 + 16, lane);
        acc0 = wmma_bf16(a, b0, acc0);
        acc1 = wmma_bf16(a, b1, acc1);
    }

    const float BNS = rsqrtf(1.f + 1e-5f);
#pragma unroll
    for (int r = 0; r < 8; ++r) {
        int row = mT * 64 + mBase + hi * 8 + r;
        if (row < COUT_G) {
            int o = g * COUT_G + row;
            float v0 = acc0[r], v1 = acc1[r];
            if (MODE == 0) {
                float s = p0[o] * BNS, t = p1[o];
                v0 = fmaxf(0.f, v0 * s + t); v1 = fmaxf(0.f, v1 * s + t);
            } else if (MODE == 1) {
                float t = p0[o]; v0 += t; v1 += t;
            } else {
                float s = p0[o] * BNS, t = p1[o];
                v0 = v0 * s + t; v1 = v1 * s + t;
            }
            size_t obase = ((size_t)b * COUT_TOT + o) * HW_ + pxBase;
            outp[obase + waveN * 32 + lo]      = v0;
            outp[obase + waveN * 32 + 16 + lo] = v1;
        }
    }
}

// ---------------------------------------------------------------------------
// Kernel 5: GroupNorm statistics over (9 channels x HW) per (b, group of 32)
// ---------------------------------------------------------------------------
__global__ void gn_stats(const float* __restrict__ wr, float* __restrict__ gnm,
                         float* __restrict__ gnr) {
    int grp = blockIdx.x, b = blockIdx.y, tid = threadIdx.x;
    const float* base = wr + ((size_t)b * 288 + grp * 9) * HW_;
    float s = 0.f, sq = 0.f;
    for (int i = tid; i < 9 * HW_; i += 256) { float v = base[i]; s += v; sq += v * v; }
    __shared__ float ls[256], lq[256];
    ls[tid] = s; lq[tid] = sq; __syncthreads();
    for (int st = 128; st > 0; st >>= 1) {
        if (tid < st) { ls[tid] += ls[tid + st]; lq[tid] += lq[tid + st]; }
        __syncthreads();
    }
    if (tid == 0) {
        float inv = 1.f / (9.f * (float)HW_);
        float m = ls[0] * inv;
        float var = lq[0] * inv - m * m;
        gnm[b * 32 + grp] = m;
        gnr[b * 32 + grp] = rsqrtf(var + 1e-5f);
    }
}

// ---------------------------------------------------------------------------
// Kernel 6: per-pixel dynamic local convolution + BN + ReLU -> y
// ---------------------------------------------------------------------------
__global__ void localconv(const float* __restrict__ v, const float* __restrict__ wr,
                          const float* __restrict__ gnm, const float* __restrict__ gnr,
                          const float* __restrict__ gng, const float* __restrict__ gnb,
                          const float* __restrict__ bng, const float* __restrict__ bnb,
                          float* __restrict__ y) {
    int tid = threadIdx.x;
    int px = blockIdx.x * 224 + tid;   // 14 * 224 = 3136
    int c = blockIdx.y, b = blockIdx.z;
    int h = px / W_, w = px % W_;
    int wch = (c >> 7) * 144 + ((c & 127) >> 3) * 9;
    int gidx = b * 32 + wch / 9;
    float m = gnm[gidx], rs = gnr[gidx];
    const float* vb = v + ((size_t)b * C_ + c) * HW_;
    const float* wb = wr + ((size_t)b * 288 + wch) * HW_ + px;
    float acc = 0.f;
#pragma unroll
    for (int t = 0; t < 9; ++t) {
        int hh = h + t / 3 - 1, ww = w + t % 3 - 1;
        float vv = (hh >= 0 && hh < H_ && ww >= 0 && ww < W_) ? vb[hh * W_ + ww] : 0.f;
        float wv = (wb[(size_t)t * HW_] - m) * rs * gng[wch + t] + gnb[wch + t];
        acc += vv * wv;
    }
    const float BNS = rsqrtf(1.f + 1e-5f);
    y[((size_t)b * C_ + c) * HW_ + px] = fmaxf(0.f, acc * (bng[c] * BNS) + bnb[c]);
}

// ---------------------------------------------------------------------------
// Kernel 7: GAP of (y + k) over HW per (b, c)   (deterministic tree reduce)
// ---------------------------------------------------------------------------
__global__ void gap_kernel(const float* __restrict__ y, const float* __restrict__ k,
                           float* __restrict__ gap) {
    int c = blockIdx.x, b = blockIdx.y, tid = threadIdx.x;
    const float* yb = y + ((size_t)b * C_ + c) * HW_;
    const float* kb = k + ((size_t)b * C_ + c) * HW_;
    float s = 0.f;
    for (int i = tid; i < HW_; i += 256) s += yb[i] + kb[i];
    __shared__ float ls[256];
    ls[tid] = s; __syncthreads();
    for (int st = 128; st > 0; st >>= 1) {
        if (tid < st) ls[tid] += ls[tid + st];
        __syncthreads();
    }
    if (tid == 0) gap[b * C_ + c] = ls[0] / (float)HW_;
}

// ---------------------------------------------------------------------------
// Kernel 8: SE MLP + radix-2 softmax -> att[b, c, 2]
// ---------------------------------------------------------------------------
__global__ void se_kernel(const float* __restrict__ gap, const float* __restrict__ w1,
                          const float* __restrict__ b1, const float* __restrict__ sg,
                          const float* __restrict__ sb, const float* __restrict__ w2,
                          const float* __restrict__ b2, float* __restrict__ att) {
    int b = blockIdx.x, tid = threadIdx.x;
    __shared__ float gs[256];
    __shared__ float ts[128];
    gs[tid] = gap[b * C_ + tid];
    __syncthreads();
    if (tid < 128) {
        float acc = b1[tid];
        for (int c = 0; c < 256; ++c) acc += gs[c] * w1[tid * 256 + c];
        const float BNS = rsqrtf(1.f + 1e-5f);
        ts[tid] = fmaxf(0.f, acc * (sg[tid] * BNS) + sb[tid]);
    }
    __syncthreads();
    float z0 = b2[2 * tid], z1 = b2[2 * tid + 1];
    for (int i = 0; i < 128; ++i) {
        float t = ts[i];
        z0 += t * w2[(2 * tid) * 128 + i];
        z1 += t * w2[(2 * tid + 1) * 128 + i];
    }
    float m = fmaxf(z0, z1);
    float e0 = expf(z0 - m), e1 = expf(z1 - m);
    float inv = 1.f / (e0 + e1);
    att[(b * C_ + tid) * 2]     = e0 * inv;
    att[(b * C_ + tid) * 2 + 1] = e1 * inv;
}

// ---------------------------------------------------------------------------
// Kernel 9: out = y * a0 + k * a1
// ---------------------------------------------------------------------------
__global__ void final_kernel(const float* __restrict__ y, const float* __restrict__ k,
                             const float* __restrict__ att, float* __restrict__ out) {
    size_t i = (size_t)blockIdx.x * 256 + threadIdx.x;
    int bc = (int)(i / HW_);
    float a0 = att[bc * 2], a1 = att[bc * 2 + 1];
    out[i] = y[i] * a0 + k[i] * a1;
}

// ---------------------------------------------------------------------------
extern "C" void kernel_launch(void* const* d_in, const int* in_sizes, int n_in,
                              void* d_out, int out_size, void* d_ws, size_t ws_size,
                              hipStream_t stream) {
    const float* x       = (const float*)d_in[0];
    const float* ke_w    = (const float*)d_in[1];
    const float* ke_g    = (const float*)d_in[2];
    const float* ke_b    = (const float*)d_in[3];
    const float* em1_w   = (const float*)d_in[4];
    const float* em1_g   = (const float*)d_in[5];
    const float* em1_b   = (const float*)d_in[6];
    const float* em2_w   = (const float*)d_in[7];
    const float* em2_bias= (const float*)d_in[8];
    const float* gn_g    = (const float*)d_in[9];
    const float* gn_b    = (const float*)d_in[10];
    const float* c1_w    = (const float*)d_in[11];
    const float* c1_g    = (const float*)d_in[12];
    const float* c1_b    = (const float*)d_in[13];
    const float* bn_g    = (const float*)d_in[14];
    const float* bn_b    = (const float*)d_in[15];
    const float* se1_w   = (const float*)d_in[16];
    const float* se1_bias= (const float*)d_in[17];
    const float* se_g    = (const float*)d_in[18];
    const float* se_b    = (const float*)d_in[19];
    const float* se2_w   = (const float*)d_in[20];
    const float* se2_bias= (const float*)d_in[21];
    float* out = (float*)d_out;

    size_t off = 0;
    auto alloc = [&](size_t nfloats) -> float* {
        float* p = (float*)((char*)d_ws + off);
        off = (off + nfloats * sizeof(float) + 255) & ~(size_t)255;
        return p;
    };
    float* kbuf = alloc((size_t)B_ * C_ * HW_);        // k
    float* e1   = alloc((size_t)B_ * 128 * HW_);       // em1 out
    float* wrb  = alloc((size_t)B_ * 288 * HW_);       // em2 out (pre-GN)
    float* vbuf = alloc((size_t)B_ * C_ * HW_);        // value
    float* ybuf = alloc((size_t)B_ * C_ * HW_);        // y
    float* gnm  = alloc(512);
    float* gnr  = alloc(512);
    float* gap  = alloc(4096);
    float* att  = alloc(8192);

    // 1. key embed: grouped 3x3 conv + BN + ReLU
    ke_conv3x3<<<dim3(49, 8, 16), 256, 0, stream>>>(x, ke_w, ke_g, ke_b, kbuf);
    // 2. em1: qk -> 128 (groups=2, 256->64), BN+ReLU
    conv1x1_wmma<256, 64, 0><<<dim3(49, 1, 32), 256, 0, stream>>>(
        x, kbuf, em1_w, em1_g, em1_b, e1);
    // 3. em2: 128 -> 288 (groups=2, 64->144), +bias
    conv1x1_wmma<64, 144, 1><<<dim3(49, 3, 32), 256, 0, stream>>>(
        e1, nullptr, em2_w, em2_bias, nullptr, wrb);
    // 4. GroupNorm stats (32 groups x 9 ch)
    gn_stats<<<dim3(32, 16), 256, 0, stream>>>(wrb, gnm, gnr);
    // 5. value: 256 -> 256 (groups=2, 128->128), BN
    conv1x1_wmma<128, 128, 2><<<dim3(49, 2, 32), 256, 0, stream>>>(
        x, nullptr, c1_w, c1_g, c1_b, vbuf);
    // 6. local conv with per-pixel dynamic kernels (+GN, +BN, +ReLU)
    localconv<<<dim3(14, 256, 16), 224, 0, stream>>>(
        vbuf, wrb, gnm, gnr, gn_g, gn_b, bn_g, bn_b, ybuf);
    // 7. GAP of (y + k)
    gap_kernel<<<dim3(256, 16), 256, 0, stream>>>(ybuf, kbuf, gap);
    // 8. SE attention
    se_kernel<<<dim3(16), 256, 0, stream>>>(gap, se1_w, se1_bias, se_g, se_b,
                                            se2_w, se2_bias, att);
    // 9. combine
    final_kernel<<<dim3(50176), 256, 0, stream>>>(ybuf, kbuf, att, out);
    (void)in_sizes; (void)n_in; (void)out_size; (void)ws_size;
}
